// NeuronFDAttentionBase_48954037240350
// MI455X (gfx1250) — matrix operand order repaired and checked by reading, not compiled, not run on hardware
//
#include <hip/hip_runtime.h>
#include <hip/hip_bf16.h>
#include <stdint.h>

typedef __bf16 bf16_t;
typedef __attribute__((ext_vector_type(16))) __bf16 v16bf;
typedef __attribute__((ext_vector_type(8)))  __bf16 v8bf;
typedef __attribute__((ext_vector_type(4)))  __bf16 v4bf;
typedef __attribute__((ext_vector_type(8)))  float  v8f;

#define S_LEN   4096
#define HID     4096
#define NQCOLS  512      // N_HEADS * HEAD_DIM
#define DHEAD   128
#define NHEADS  4
#define LTS     40       // padded LDS row stride for 32-wide tiles (80B, 16B aligned, bank-conflict-free)
#define KSTR    136      // padded LDS row stride for 128-wide K tile (272B)

__device__ __forceinline__ v16bf mk16(v8bf lo, v8bf hi) {
  v16bf r;
#pragma unroll
  for (int i = 0; i < 8; ++i) { r[i] = lo[i]; r[i + 8] = hi[i]; }
  return r;
}

__device__ __forceinline__ v8f wmma_bf16(v16bf a, v16bf b, v8f c) {
  return __builtin_amdgcn_wmma_f32_16x16x32_bf16(false, a, false, b, (short)0, c,
                                                 false, false);
}

__device__ __forceinline__ uint32_t pack2(bf16_t a, bf16_t b) {
  union { bf16_t h[2]; uint32_t u; } x;
  x.h[0] = a; x.h[1] = b;
  return x.u;
}

// ---------------------------------------------------------------- conversion
__global__ __launch_bounds__(256) void cvt_f32_to_bf16(const float* __restrict__ in,
                                                       bf16_t* __restrict__ out,
                                                       int n4) {
  int i = blockIdx.x * 256 + threadIdx.x;
  if (i >= n4) return;
  const float4 v = reinterpret_cast<const float4*>(in)[i];
  v4bf o;
  o[0] = (bf16_t)v.x; o[1] = (bf16_t)v.y; o[2] = (bf16_t)v.z; o[3] = (bf16_t)v.w;
  reinterpret_cast<v4bf*>(out)[i] = o;
}

// ------------------------------------------------------------- generic GEMM
// C(MxN) = A(MxK) * B(KxN), both bf16 row-major, fp32 accum.
// Compile-time N/K -> constant strides, branch-free epilogue.
// Block 128x128 (8 waves, each 32Mx64N). Double-buffered LDS + register
// prefetch of the next K-tile so global latency overlaps the WMMAs.
template <int N, int K, bool OUTF32>
__global__ __launch_bounds__(256) void gemm_bf16_nn(const bf16_t* __restrict__ A,
                                                    const bf16_t* __restrict__ B,
                                                    void* __restrict__ Cv) {
  __shared__ bf16_t Asl[2][128 * LTS];   // [m][k] k-contiguous
  __shared__ bf16_t Btl[2][128 * LTS];   // [n][k] transposed, k-contiguous
  const int mb = blockIdx.x, nb = blockIdx.y;
  const int tid = threadIdx.x;
  const int lane = tid & 31, wave = tid >> 5;
  const int wm = wave & 3, wn = wave >> 2;  // wave grid 4(M) x 2(N)
  const int lm = lane & 15, lh = lane >> 4;

  // loader mapping (256 threads)
  const int arow = tid >> 1, acol = (tid & 1) * 16;      // A: 128 rows x 32k
  const int kpr  = tid >> 4, nc   = (tid & 15) * 8;      // B: 16 k-pairs x 128n
  const bf16_t* aptr = A + (size_t)(mb * 128 + arow) * K + acol;
  const bf16_t* bptr = B + (size_t)(2 * kpr) * N + nb * 128 + nc;

  v8bf ra0, ra1, rb0, rb1;
  auto load_regs = [&](int kb) {
    ra0 = *(const v8bf*)(aptr + kb);
    ra1 = *(const v8bf*)(aptr + kb + 8);
    rb0 = *(const v8bf*)(bptr + (size_t)kb * N);
    rb1 = *(const v8bf*)(bptr + (size_t)(kb + 1) * N);
  };
  auto store_lds = [&](int buf) {
    *(v8bf*)(&Asl[buf][arow * LTS + acol])     = ra0;
    *(v8bf*)(&Asl[buf][arow * LTS + acol + 8]) = ra1;
#pragma unroll
    for (int i = 0; i < 8; ++i)
      *(uint32_t*)(&Btl[buf][(nc + i) * LTS + 2 * kpr]) = pack2(rb0[i], rb1[i]);
  };

  const v8f vzero = {};
  v8f acc[2][4];
#pragma unroll
  for (int mt = 0; mt < 2; ++mt)
#pragma unroll
    for (int nt = 0; nt < 4; ++nt) acc[mt][nt] = vzero;

  load_regs(0);
  store_lds(0);
  __syncthreads();

  for (int kb = 0; kb < K; kb += 32) {
    const int cur = (kb >> 5) & 1;
    const bool more = (kb + 32) < K;
    if (more) load_regs(kb + 32);   // global prefetch overlaps WMMAs below

    v16bf afr[2], bfr[4];
#pragma unroll
    for (int mt = 0; mt < 2; ++mt) {
      const bf16_t* p = &Asl[cur][(wm * 32 + mt * 16 + lm) * LTS];
      afr[mt] = mk16(*(const v8bf*)(p + lh * 8), *(const v8bf*)(p + 16 + lh * 8));
    }
#pragma unroll
    for (int nt = 0; nt < 4; ++nt) {
      const bf16_t* p = &Btl[cur][(wn * 64 + nt * 16 + lm) * LTS + lh * 8];
      bfr[nt] = mk16(*(const v8bf*)(p), *(const v8bf*)(p + 16));
    }
#pragma unroll
    for (int mt = 0; mt < 2; ++mt)
#pragma unroll
      for (int nt = 0; nt < 4; ++nt)
        acc[mt][nt] = wmma_bf16(afr[mt], bfr[nt], acc[mt][nt]);

    if (more) store_lds(1 - cur);   // safe: peers only read [cur] this iter
    __syncthreads();
  }

  // branch-free epilogue, compile-time stride N
#pragma unroll
  for (int mt = 0; mt < 2; ++mt)
#pragma unroll
    for (int nt = 0; nt < 4; ++nt) {
      const int row = mb * 128 + wm * 32 + mt * 16 + 8 * lh;
      const int col = nb * 128 + wn * 64 + nt * 16 + lm;
      if (OUTF32) {
        float* C = (float*)Cv + (size_t)row * N + col;
#pragma unroll
        for (int r = 0; r < 8; ++r) C[(size_t)r * N] = acc[mt][nt][r];
      } else {
        bf16_t* C = (bf16_t*)Cv + (size_t)row * N + col;
#pragma unroll
        for (int r = 0; r < 8; ++r) C[(size_t)r * N] = (bf16_t)acc[mt][nt][r];
      }
    }
}

// ---------------------------------------------------------- flash attention
// grid = (S/128 query blocks, NHEADS). Each wave owns 16 query rows.
// Double-buffered K/V tiles with register prefetch.
__global__ __launch_bounds__(256) void flash_attn_causal(
    const bf16_t* __restrict__ Q,    // [S][512]
    const bf16_t* __restrict__ Km,   // [S][128]
    const bf16_t* __restrict__ Vm,   // [S][128]
    bf16_t* __restrict__ Oatt) {     // [S][512]
  const int qb = blockIdx.x, head = blockIdx.y;
  const int tid = threadIdx.x;
  const int lane = tid & 31, wave = tid >> 5;
  const int lm = lane & 15, lh = lane >> 4;
  const float scale = 0.08838834764831845f;  // 1/sqrt(128)

  __shared__ bf16_t Ksl[2][32 * KSTR];     // [key][d] row-major
  __shared__ bf16_t Vtl[2][128 * LTS];     // [d][key] transposed
  __shared__ bf16_t Psl[8][16 * LTS];      // per-wave P tile [row][key]

  // loader mapping
  const int krow = tid >> 3, kc = (tid & 7) * 16;   // K: 32 rows x 128d
  const int rpr  = tid >> 4, vc = (tid & 15) * 8;   // V: 16 key-pairs x 128d
  const bf16_t* kptr = Km + (size_t)krow * DHEAD + kc;
  const bf16_t* vptr = Vm + (size_t)(2 * rpr) * DHEAD + vc;

  v8bf rk0, rk1, rv0, rv1;
  auto load_regs = [&](int kb) {
    rk0 = *(const v8bf*)(kptr + (size_t)kb * DHEAD);
    rk1 = *(const v8bf*)(kptr + (size_t)kb * DHEAD + 8);
    rv0 = *(const v8bf*)(vptr + (size_t)kb * DHEAD);
    rv1 = *(const v8bf*)(vptr + (size_t)(kb + 1) * DHEAD);
  };
  auto store_lds = [&](int buf) {
    *(v8bf*)(&Ksl[buf][krow * KSTR + kc])     = rk0;
    *(v8bf*)(&Ksl[buf][krow * KSTR + kc + 8]) = rk1;
#pragma unroll
    for (int i = 0; i < 8; ++i)
      *(uint32_t*)(&Vtl[buf][(vc + i) * LTS + 2 * rpr]) = pack2(rv0[i], rv1[i]);
  };

  // cache Q fragments (4 x K=32 chunks over d=128)
  const int qrow = qb * 128 + wave * 16 + lm;
  const bf16_t* qptr = Q + (size_t)qrow * NQCOLS + head * DHEAD;
  v16bf qf[4];
#pragma unroll
  for (int c = 0; c < 4; ++c)
    qf[c] = mk16(*(const v8bf*)(qptr + c * 32 + lh * 8),
                 *(const v8bf*)(qptr + c * 32 + 16 + lh * 8));

  const v8f vzero = {};
  v8f o[8];
#pragma unroll
  for (int d = 0; d < 8; ++d) o[d] = vzero;
  float mrow[8], lrow[8];
#pragma unroll
  for (int r = 0; r < 8; ++r) { mrow[r] = -1e30f; lrow[r] = 0.0f; }

  const int kend = (qb + 1) * 128;
  load_regs(0);
  store_lds(0);
  __syncthreads();

  for (int kb = 0; kb < kend; kb += 32) {
    const int cur = (kb >> 5) & 1;
    const bool more = (kb + 32) < kend;
    if (more) load_regs(kb + 32);

    // S = Q * K^T : two 16x16 tiles (keys kb..+15, kb+16..+31)
    v8f sfr[2];
#pragma unroll
    for (int t = 0; t < 2; ++t) {
      v8f s = vzero;
#pragma unroll
      for (int c = 0; c < 4; ++c) {
        const bf16_t* p = &Ksl[cur][(t * 16 + lm) * KSTR + c * 32 + lh * 8];
        v16bf bf = mk16(*(const v8bf*)(p), *(const v8bf*)(p + 16));
        s = wmma_bf16(qf[c], bf, s);
      }
      sfr[t] = s;
    }

    // scale + causal mask + row max (rows r+8h live on this lane)
    float rmax[8];
#pragma unroll
    for (int r = 0; r < 8; ++r) {
      int qg = qb * 128 + wave * 16 + r + 8 * lh;
      float s0 = sfr[0][r] * scale; if (kb + lm > qg)      s0 = -1e30f;
      float s1 = sfr[1][r] * scale; if (kb + 16 + lm > qg) s1 = -1e30f;
      sfr[0][r] = s0; sfr[1][r] = s1;
      rmax[r] = fmaxf(s0, s1);
    }
#pragma unroll
    for (int m = 1; m < 16; m <<= 1)
#pragma unroll
      for (int r = 0; r < 8; ++r)
        rmax[r] = fmaxf(rmax[r], __shfl_xor(rmax[r], m, 32));

    float alpha[8], rsum[8];
#pragma unroll
    for (int r = 0; r < 8; ++r) {
      float mn = fmaxf(mrow[r], rmax[r]);
      alpha[r] = __expf(mrow[r] - mn);
      mrow[r] = mn;
      float p0 = __expf(sfr[0][r] - mn);
      float p1 = __expf(sfr[1][r] - mn);
      sfr[0][r] = p0; sfr[1][r] = p1;
      rsum[r] = p0 + p1;
    }
#pragma unroll
    for (int m = 1; m < 16; m <<= 1)
#pragma unroll
      for (int r = 0; r < 8; ++r)
        rsum[r] += __shfl_xor(rsum[r], m, 32);
#pragma unroll
    for (int r = 0; r < 8; ++r) lrow[r] = lrow[r] * alpha[r] + rsum[r];
#pragma unroll
    for (int d = 0; d < 8; ++d)
#pragma unroll
      for (int r = 0; r < 8; ++r) o[d][r] *= alpha[r];

    // P (D-frag layout) -> per-wave LDS -> A-frag layout
#pragma unroll
    for (int t = 0; t < 2; ++t)
#pragma unroll
      for (int r = 0; r < 8; ++r)
        Psl[wave][(r + 8 * lh) * LTS + t * 16 + lm] = (bf16_t)sfr[t][r];

    const bf16_t* pp = &Psl[wave][lm * LTS];
    v16bf pf = mk16(*(const v8bf*)(pp + lh * 8), *(const v8bf*)(pp + 16 + lh * 8));

    // O += P * V
#pragma unroll
    for (int d = 0; d < 8; ++d) {
      const bf16_t* vp = &Vtl[cur][(d * 16 + lm) * LTS + lh * 8];
      v16bf vf = mk16(*(const v8bf*)(vp), *(const v8bf*)(vp + 16));
      o[d] = wmma_bf16(pf, vf, o[d]);
    }

    if (more) store_lds(1 - cur);   // peers only read [cur] this iteration
    __syncthreads();
  }

  // normalize + write (compile-time stride)
#pragma unroll
  for (int d = 0; d < 8; ++d) {
    const int row = qb * 128 + wave * 16 + 8 * lh;
    const int col = head * DHEAD + d * 16 + lm;
    bf16_t* op = Oatt + (size_t)row * NQCOLS + col;
#pragma unroll
    for (int r = 0; r < 8; ++r)
      op[(size_t)r * NQCOLS] = (bf16_t)(o[d][r] / lrow[r]);
  }
}

// ----------------------------------------------------------- K/V selection
__global__ __launch_bounds__(256) void select_kv(const bf16_t* __restrict__ Kb,
                                                 const bf16_t* __restrict__ Vb,
                                                 float* __restrict__ out,
                                                 const int* __restrict__ ncg) {
  int stride = (ncg != nullptr) ? ncg[0] : 8;
  if (stride <= 0) stride = 8;
  const int nsel = S_LEN / stride;
  int i = blockIdx.x * 256 + threadIdx.x;
  if (i >= nsel * DHEAD) return;
  int s = i >> 7, d = i & 127;
  const size_t base = (size_t)S_LEN * HID;  // after attn_output
  out[base + i]                        = (float)Kb[(size_t)(s * stride) * DHEAD + d];
  out[base + (size_t)nsel * DHEAD + i] = (float)Vb[(size_t)(s * stride) * DHEAD + d];
}

// ---------------------------------------------------------------- launcher
extern "C" void kernel_launch(void* const* d_in, const int* in_sizes, int n_in,
                              void* d_out, int out_size, void* d_ws, size_t ws_size,
                              hipStream_t stream) {
  const float* hidden = (const float*)d_in[0];
  const float* Wq     = (const float*)d_in[1];
  const float* Wk     = (const float*)d_in[2];
  const float* Wv     = (const float*)d_in[3];
  const float* Wo     = (const float*)d_in[4];
  const int*   ncg    = (n_in > 5) ? (const int*)d_in[5] : nullptr;
  float* out = (float*)d_out;

  char* ws = (char*)d_ws;
  size_t off = 0;
  bf16_t* hb  = (bf16_t*)(ws + off); off += (size_t)S_LEN * HID * 2;      // 32MB
  bf16_t* wqb = (bf16_t*)(ws + off); off += (size_t)HID * NQCOLS * 2;     // 4MB
  bf16_t* wkb = (bf16_t*)(ws + off); off += (size_t)HID * DHEAD * 2;      // 1MB
  bf16_t* wvb = (bf16_t*)(ws + off); off += (size_t)HID * DHEAD * 2;      // 1MB
  bf16_t* wob = (bf16_t*)(ws + off); off += (size_t)NQCOLS * HID * 2;     // 4MB
  bf16_t* qb  = (bf16_t*)(ws + off); off += (size_t)S_LEN * NQCOLS * 2;   // 4MB
  bf16_t* kb  = (bf16_t*)(ws + off); off += (size_t)S_LEN * DHEAD * 2;    // 1MB
  bf16_t* vb  = (bf16_t*)(ws + off); off += (size_t)S_LEN * DHEAD * 2;    // 1MB
  bf16_t* ab  = (bf16_t*)(ws + off); off += (size_t)S_LEN * NQCOLS * 2;   // 4MB

  // 1) fp32 -> bf16
  auto cvt = [&](const float* src, bf16_t* dst, size_t n) {
    int n4 = (int)(n / 4);
    cvt_f32_to_bf16<<<(n4 + 255) / 256, 256, 0, stream>>>(src, dst, n4);
  };
  cvt(hidden, hb,  (size_t)S_LEN * HID);
  cvt(Wq,     wqb, (size_t)HID * NQCOLS);
  cvt(Wk,     wkb, (size_t)HID * DHEAD);
  cvt(Wv,     wvb, (size_t)HID * DHEAD);
  cvt(Wo,     wob, (size_t)NQCOLS * HID);

  // 2) projections (bf16 out)
  gemm_bf16_nn<NQCOLS, HID, false>
      <<<dim3(S_LEN / 128, NQCOLS / 128), 256, 0, stream>>>(hb, wqb, qb);
  gemm_bf16_nn<DHEAD, HID, false>
      <<<dim3(S_LEN / 128, DHEAD / 128), 256, 0, stream>>>(hb, wkb, kb);
  gemm_bf16_nn<DHEAD, HID, false>
      <<<dim3(S_LEN / 128, DHEAD / 128), 256, 0, stream>>>(hb, wvb, vb);

  // 3) causal flash attention
  flash_attn_causal<<<dim3(S_LEN / 128, NHEADS), 256, 0, stream>>>(qb, kb, vb, ab);

  // 4) output projection -> fp32 d_out
  gemm_bf16_nn<HID, NQCOLS, true>
      <<<dim3(S_LEN / 128, HID / 128), 256, 0, stream>>>(ab, wob, out);

  // 5) K_sel / V_sel
  select_kv<<<(S_LEN * DHEAD + 255) / 256, 256, 0, stream>>>(kb, vb, out, ncg);
}
